// EncoderLayer_78855599555051
// MI455X (gfx1250) — compile-verified, matrix operand order checked by reading, not run on hardware
//
#include <hip/hip_runtime.h>
#include <hip/hip_bf16.h>
#include <math.h>

// ---------------------------------------------------------------------------
// Types for CDNA5 WMMA
// ---------------------------------------------------------------------------
typedef __attribute__((ext_vector_type(16))) _Float16 v16h;
typedef __attribute__((ext_vector_type(8)))  _Float16 v8h;
typedef __attribute__((ext_vector_type(8)))  float    v8f;

#define GN 50000
#define GE 800000
#define GD 128
#define GH 8
#define GC 16
#define GEDIM 16
#define GDFF 512

// ---------------------------------------------------------------------------
// Tiled WMMA GEMM: C[M,N] = act(A[M,K] @ B[K,N] + bias)
// f32 in/out, f16 operands, f32 accumulate via v_wmma_f32_16x16x32_f16.
//
// Block = 128 threads = 4 waves. Block tile 128x64. Each wave owns TWO 16-row
// tiles and iterates 4 column tiles of 16: 8 WMMAs per k-step per wave.
//
// LDS holds A and B^T tiles in *fragment order*: for wave-tile wt, lane L
// stores its 16 fragment f16 elements contiguously at
//     addr_f16 = wt*512 + L*16 + i
// so the compute phase is pure ds_load_b128 (no scalar gathers).
// Fragment element mapping (ISA 7.12.2, 16-bit A/B 16x32):
//     K(L, i) = (i%8) + (i/8)*16 + (L/16)*8 ,  row/col = L%16
// A K-group of 8 consecutive k (g = k/8) maps to contiguous elements
// ibase=(g/2)*8.. of lane half=(g&1): one 16B LDS store per group, and the
// matching global read is two aligned float4 (global_load_b128) per group.
//
// Requires N%64==0, K%32==0 (true here: N in {128,512}, K in {128,512}).
// ---------------------------------------------------------------------------
#define TM 128
#define TN 64
#define TK 32

__global__ __launch_bounds__(128) void wmma_gemm_f16(
    const float* __restrict__ A, const float* __restrict__ B,
    const float* __restrict__ bias, float* __restrict__ C,
    int M, int N, int K, int act /*0=none, 1=relu*/)
{
    // fragment-order staging buffers: (TM/16) and (TN/16) tiles of 512 f16
    __shared__ __align__(32) _Float16 sA[(TM / 16) * 512];   // 8 KB
    __shared__ __align__(32) _Float16 sBt[(TN / 16) * 512];  // 4 KB

    const int bm   = blockIdx.x * TM;
    const int bn   = blockIdx.y * TN;
    const int tid  = threadIdx.x;
    const int wave = tid >> 5;
    const int lane = tid & 31;

    v8f acc0[4] = {};
    v8f acc1[4] = {};

    for (int k0 = 0; k0 < K; k0 += TK) {
        // ---- Stage A tile (TM x 32) into fragment order -------------------
        // 512 groups of 8 consecutive k; exactly 4 per thread (unrolled).
        // Guard hoisted per group -> vector float4 loads, one 16B LDS store.
        #pragma unroll
        for (int u = 0; u < (TM * TK) / 8 / 128; ++u) {
            int G = tid + u * 128;
            int r = G >> 2, g = G & 3;
            int gr = bm + r;
            v8h h;
            if (gr < M) {
                const float* ap = A + (size_t)gr * K + k0 + (g << 3);
                float4 lo = *(const float4*)(ap);
                float4 hi = *(const float4*)(ap + 4);
                h[0] = (_Float16)lo.x; h[1] = (_Float16)lo.y;
                h[2] = (_Float16)lo.z; h[3] = (_Float16)lo.w;
                h[4] = (_Float16)hi.x; h[5] = (_Float16)hi.y;
                h[6] = (_Float16)hi.z; h[7] = (_Float16)hi.w;
            } else {
                #pragma unroll
                for (int j = 0; j < 8; ++j) h[j] = (_Float16)0.0f;
            }
            int half  = g & 1;
            int ibase = (g >> 1) << 3;
            int l16   = (half << 4) + (r & 15);
            int wt    = r >> 4;
            *(v8h*)&sA[(wt << 9) + (l16 << 4) + ibase] = h;
        }
        // ---- Stage B tile (32 x TN) transposed into fragment order --------
        // 256 groups; exactly 2 per thread. Per-j reads are coalesced across
        // threads (consecutive n for fixed k); B is L2-resident weight data.
        #pragma unroll
        for (int u = 0; u < (TK * TN) / 8 / 128; ++u) {
            int G = tid + u * 128;
            int kg = G >> 6, n = G & 63;
            const float* bp = B + (size_t)(k0 + (kg << 3)) * N + bn + n;
            v8h h;
            #pragma unroll
            for (int j = 0; j < 8; ++j)
                h[j] = (_Float16)bp[(size_t)j * N];
            int half  = kg & 1;
            int ibase = (kg >> 1) << 3;
            int l16   = (half << 4) + (n & 15);
            int ct    = n >> 4;
            *(v8h*)&sBt[(ct << 9) + (l16 << 4) + ibase] = h;
        }
        __syncthreads();

        // ---- Compute: vector fragment loads + 8 WMMAs ---------------------
        const int fo = lane << 4;
        v16h a0 = *(const v16h*)&sA[((wave << 1) << 9) + fo];
        v16h a1 = *(const v16h*)&sA[(((wave << 1) + 1) << 9) + fo];
        #pragma unroll
        for (int ct = 0; ct < 4; ++ct) {
            v16h b = *(const v16h*)&sBt[(ct << 9) + fo];
            acc0[ct] = __builtin_amdgcn_wmma_f32_16x16x32_f16(
                false, a0, false, b, (short)0, acc0[ct], false, false);
            acc1[ct] = __builtin_amdgcn_wmma_f32_16x16x32_f16(
                false, a1, false, b, (short)0, acc1[ct], false, false);
        }
        __syncthreads();
    }

    // D layout: VGPR r -> row = 8*(lane/16) + r ; lane%16 -> col
    const int lrow  = lane & 15;
    const int half8 = (lane >> 4) << 3;
    #pragma unroll
    for (int t = 0; t < 2; ++t) {
        const v8f* accp = t ? acc1 : acc0;
        const int rbase = bm + ((wave << 1) + t) * 16 + half8;
        #pragma unroll
        for (int ct = 0; ct < 4; ++ct) {
            const int col = bn + (ct << 4) + lrow;
            const float bv = bias ? bias[col] : 0.0f;
            #pragma unroll
            for (int r = 0; r < 8; ++r) {
                int row = rbase + r;
                if (row < M) {
                    float v = accp[ct][r] + bv;
                    if (act == 1) v = fmaxf(v, 0.0f);
                    C[(size_t)row * N + col] = v;
                }
            }
        }
    }
}

// ---------------------------------------------------------------------------
// Per-node head scores: s_src[n,h] = sum_c xh[n,h*16+c]*a_src[h,c] (and dst)
// ---------------------------------------------------------------------------
__global__ void head_scores(const float* __restrict__ xh,
                            const float* __restrict__ a_src,
                            const float* __restrict__ a_dst,
                            float* __restrict__ ssrc, float* __restrict__ sdst,
                            int n_nodes)
{
    int idx = blockIdx.x * blockDim.x + threadIdx.x;
    if (idx >= n_nodes * GH) return;
    int n = idx >> 3, h = idx & 7;
    const float* xr = xh + (size_t)n * GD + h * GC;
    float s1 = 0.f, s2 = 0.f;
    #pragma unroll
    for (int c = 0; c < GC; ++c) {
        float v = xr[c];
        s1 += v * a_src[h * GC + c];
        s2 += v * a_dst[h * GC + c];
    }
    ssrc[idx] = s1;
    sdst[idx] = s2;
}

// ---------------------------------------------------------------------------
// Fold We (16x128) and a_e (8x16) into Wae (16x8):
// Wae[k,h] = sum_c We[k, h*16+c] * a_e[h,c]
// ---------------------------------------------------------------------------
__global__ void wae_precompute(const float* __restrict__ We,
                               const float* __restrict__ ae,
                               float* __restrict__ wae)
{
    int idx = threadIdx.x;      // 128 threads: k in [0,16), h in [0,8)
    int k = idx >> 3, h = idx & 7;
    float s = 0.f;
    #pragma unroll
    for (int c = 0; c < GC; ++c)
        s += We[(size_t)k * GD + h * GC + c] * ae[h * GC + c];
    wae[k * GH + h] = s;
}

// ---------------------------------------------------------------------------
// Order-preserving float<->uint keys for atomicMax segment max
// ---------------------------------------------------------------------------
__device__ __forceinline__ unsigned float_key(float f) {
    int i = __float_as_int(f);
    return (i >= 0) ? ((unsigned)i | 0x80000000u) : (unsigned)(~i);
}
__device__ __forceinline__ float key_float(unsigned k) {
    int i = (k & 0x80000000u) ? (int)(k & 0x7fffffffu) : ~(int)k;
    return __int_as_float(i);
}

// ---------------------------------------------------------------------------
// Edge scoring: alpha_raw = lrelu0.2(s_src[src]+s_dst[dst]+ ew@Wae), plus
// segment max into mkey[dst,h].
// ---------------------------------------------------------------------------
__global__ void edge_score(const int* __restrict__ ei,
                           const float* __restrict__ ew,
                           const float* __restrict__ wae,
                           const float* __restrict__ ssrc,
                           const float* __restrict__ sdst,
                           float* __restrict__ alpha,
                           unsigned* __restrict__ mkey, int E)
{
    int e = blockIdx.x * blockDim.x + threadIdx.x;
    if (e >= E) return;
    int src = ei[e];
    int dst = ei[E + e];
    float w[GEDIM];
    #pragma unroll
    for (int k = 0; k < GEDIM; ++k) w[k] = ew[(size_t)e * GEDIM + k];
    #pragma unroll
    for (int h = 0; h < GH; ++h) {
        float se = 0.f;
        #pragma unroll
        for (int k = 0; k < GEDIM; ++k) se += w[k] * wae[k * GH + h];
        float a = ssrc[src * GH + h] + sdst[dst * GH + h] + se;
        a = (a > 0.f) ? a : 0.2f * a;            // leaky_relu 0.2
        alpha[(size_t)e * GH + h] = a;
        atomicMax(&mkey[dst * GH + h], float_key(a));
    }
}

// ---------------------------------------------------------------------------
// exp(alpha - max[dst]) and partition sum z[dst,h]
// ---------------------------------------------------------------------------
__global__ void edge_exp(const int* __restrict__ ei,
                         float* __restrict__ alpha,
                         const unsigned* __restrict__ mkey,
                         float* __restrict__ z, int E)
{
    int idx = blockIdx.x * blockDim.x + threadIdx.x;
    if (idx >= E * GH) return;
    int e = idx >> 3, h = idx & 7;
    int dst = ei[E + e];
    float m = key_float(mkey[dst * GH + h]);
    float v = expf(alpha[idx] - m);
    alpha[idx] = v;
    atomicAdd(&z[dst * GH + h], v);
}

// ---------------------------------------------------------------------------
// Message + scatter: agg[dst,c] += xh[src,c] * alpha/(z+eps)
// One thread per (edge, channel).
// ---------------------------------------------------------------------------
__global__ void edge_message(const int* __restrict__ ei,
                             const float* __restrict__ alpha,
                             const float* __restrict__ z,
                             const float* __restrict__ xh,
                             float* __restrict__ agg, int E)
{
    int idx = blockIdx.x * blockDim.x + threadIdx.x;
    if (idx >= E * GD) return;
    int e = idx >> 7;
    int c = idx & 127;
    int h = c >> 4;
    int src = ei[e];
    int dst = ei[E + e];
    if (c == 0) __builtin_prefetch(xh + (size_t)src * GD, 0, 0);
    float coef = alpha[(size_t)e * GH + h] / (z[dst * GH + h] + 1e-16f);
    atomicAdd(&agg[(size_t)dst * GD + c], xh[(size_t)src * GD + c] * coef);
}

// ---------------------------------------------------------------------------
// Node finalize: y = resid + lrelu0.01( LN(x (+bias)) )
// LN with ddof=1 (divide by D-1) and eps added to sd — per reference.
// One block of 128 threads per node.
// ---------------------------------------------------------------------------
__global__ __launch_bounds__(128) void node_finalize(
    const float* __restrict__ x_in, const float* __restrict__ bias,
    const float* __restrict__ gam, const float* __restrict__ bet,
    const float* __restrict__ resid, float* __restrict__ out, int n_nodes)
{
    int n = blockIdx.x;
    int c = threadIdx.x;
    float x = x_in[(size_t)n * GD + c];
    if (bias) x += bias[c];

    __shared__ float red[GD];
    red[c] = x;
    __syncthreads();
    for (int s = 64; s > 0; s >>= 1) {
        if (c < s) red[c] += red[c + s];
        __syncthreads();
    }
    float mu = red[0] * (1.0f / GD);
    __syncthreads();
    float d = x - mu;
    red[c] = d * d;
    __syncthreads();
    for (int s = 64; s > 0; s >>= 1) {
        if (c < s) red[c] += red[c + s];
        __syncthreads();
    }
    float sd = sqrtf(red[0] * (1.0f / (GD - 1)));   // unbiased std
    float y = gam[c] * d / (sd + 1e-6f) + bet[c];
    y = (y > 0.f) ? y : 0.01f * y;                   // leaky_relu 0.01
    out[(size_t)n * GD + c] = resid[(size_t)n * GD + c] + y;
}

// ---------------------------------------------------------------------------
// Host side
// ---------------------------------------------------------------------------
static inline int cdiv(int a, int b) { return (a + b - 1) / b; }

extern "C" void kernel_launch(void* const* d_in, const int* in_sizes, int n_in,
                              void* d_out, int out_size, void* d_ws, size_t ws_size,
                              hipStream_t stream)
{
    const float* nf0 = (const float*)d_in[0];
    const int*   ei  = (const int*)  d_in[1];
    const float* ew  = (const float*)d_in[2];
    const float* W1  = (const float*)d_in[3];
    const float* as1 = (const float*)d_in[4];
    const float* ad1 = (const float*)d_in[5];
    const float* We1 = (const float*)d_in[6];
    const float* ae1 = (const float*)d_in[7];
    const float* b1  = (const float*)d_in[8];
    const float* ga1 = (const float*)d_in[9];
    const float* be1 = (const float*)d_in[10];
    const float* W2  = (const float*)d_in[11];
    const float* as2 = (const float*)d_in[12];
    const float* ad2 = (const float*)d_in[13];
    const float* We2 = (const float*)d_in[14];
    const float* ae2 = (const float*)d_in[15];
    const float* b2  = (const float*)d_in[16];
    const float* ga2 = (const float*)d_in[17];
    const float* be2 = (const float*)d_in[18];
    const float* Wf1 = (const float*)d_in[19];
    const float* bf1 = (const float*)d_in[20];
    const float* Wf2 = (const float*)d_in[21];
    const float* bf2 = (const float*)d_in[22];
    const float* ga3 = (const float*)d_in[23];
    const float* be3 = (const float*)d_in[24];

    float* ws = (float*)d_ws;
    const size_t ND = (size_t)GN * GD;   // 6.4M
    const size_t EH = (size_t)GE * GH;   // 6.4M
    size_t oXH = 0;
    size_t oAL = oXH + ND;
    size_t oAG = oAL + EH;
    size_t oNA = oAG + ND;
    size_t oNB = oNA + ND;
    size_t oFF = oNB + ND;
    size_t oSS = oFF + ND;
    size_t oSD = oSS + (size_t)GN * GH;
    size_t oMK = oSD + (size_t)GN * GH;
    size_t oZZ = oMK + (size_t)GN * GH;
    size_t oWA = oZZ + (size_t)GN * GH;

    float*    xh    = ws + oXH;
    float*    alpha = ws + oAL;
    float*    agg   = ws + oAG;
    float*    nfA   = ws + oNA;
    float*    nfB   = ws + oNB;
    float*    ffb   = ws + oFF;
    float*    ssrc  = ws + oSS;
    float*    sdst  = ws + oSD;
    unsigned* mkey  = (unsigned*)(ws + oMK);
    float*    zz    = ws + oZZ;
    float*    wae   = ws + oWA;
    // FFN hidden (N x 512) aliases [xh|alpha|agg|nfA] — all dead by then.
    float*    h1    = ws + oXH;

    auto run_gat = [&](const float* x_in, const float* W, const float* a_s,
                       const float* a_d, const float* We, const float* a_e,
                       const float* bia, const float* g, const float* be,
                       float* x_out) {
        hipMemsetAsync(agg,  0, ND * sizeof(float), stream);
        hipMemsetAsync(mkey, 0, (size_t)GN * GH * sizeof(unsigned), stream);
        hipMemsetAsync(zz,   0, (size_t)GN * GH * sizeof(float), stream);

        dim3 gg(cdiv(GN, TM), GD / TN);
        wmma_gemm_f16<<<gg, 128, 0, stream>>>(x_in, W, nullptr, xh, GN, GD, GD, 0);
        head_scores<<<cdiv(GN * GH, 256), 256, 0, stream>>>(xh, a_s, a_d, ssrc, sdst, GN);
        wae_precompute<<<1, 128, 0, stream>>>(We, a_e, wae);
        edge_score<<<cdiv(GE, 256), 256, 0, stream>>>(ei, ew, wae, ssrc, sdst, alpha, mkey, GE);
        edge_exp<<<cdiv(GE * GH, 256), 256, 0, stream>>>(ei, alpha, mkey, zz, GE);
        edge_message<<<cdiv(GE * GD, 256), 256, 0, stream>>>(ei, alpha, zz, xh, agg, GE);
        node_finalize<<<GN, 128, 0, stream>>>(agg, bia, g, be, x_in, x_out, GN);
    };

    run_gat(nf0, W1, as1, ad1, We1, ae1, b1, ga1, be1, nfA);
    run_gat(nfA, W2, as2, ad2, We2, ae2, b2, ga2, be2, nfB);

    // FFN: h1 = relu(nfB @ Wf1 + bf1);  ff = h1 @ Wf2 + bf2
    dim3 gf1(cdiv(GN, TM), GDFF / TN);
    wmma_gemm_f16<<<gf1, 128, 0, stream>>>(nfB, Wf1, bf1, h1, GN, GDFF, GD, 1);
    dim3 gf2(cdiv(GN, TM), GD / TN);
    wmma_gemm_f16<<<gf2, 128, 0, stream>>>(h1, Wf2, bf2, ffb, GN, GD, GDFF, 0);
    node_finalize<<<GN, 128, 0, stream>>>(ffb, nullptr, ga3, be3, nfB, (float*)d_out, GN);
}
